// SelfAttention_2052994367600
// MI455X (gfx1250) — compile-verified, matrix operand order checked by reading, not run on hardware
//
#include <hip/hip_runtime.h>
#include <hip/hip_bf16.h>

// ---------------------------------------------------------------------------
// Self-attention (B=8, L=2048, D=1024, 1 head) for gfx1250 (MI455X, wave32)
// Pipeline:
//   1) elementwise f32->bf16 convert of the 3 activation sources (once)
//   2) weight transpose+convert to bf16 N-major
//   3) 3x projection GEMM (bf16 WMMA, f32 acc, +bias); V stored transposed
//   4) QK^T scores (f32)  5) row softmax -> bf16 P  6) P@V -> f32 out
// All GEMMs: per-wave 64(M) x 64(N) tiles, v_wmma_f32_16x16x32_bf16,
// all 16 operand loads issued before the 16 WMMAs of each k-step.
// ---------------------------------------------------------------------------

#define BATCH 8
#define SEQ   2048
#define DIM   1024

typedef __bf16 bf16_t;
typedef bf16_t v4bf  __attribute__((ext_vector_type(4)));
typedef bf16_t v8bf  __attribute__((ext_vector_type(8)));
typedef bf16_t v16bf __attribute__((ext_vector_type(16)));
typedef float  v8f   __attribute__((ext_vector_type(8)));
typedef float  v4f   __attribute__((ext_vector_type(4)));

// round-to-nearest-even f32 -> bf16 (used in prep kernels, projection epilogue and softmax)
static __device__ __forceinline__ bf16_t f2bf(float f) {
    unsigned u = __builtin_bit_cast(unsigned, f);
    unsigned r = u + 0x7FFFu + ((u >> 16) & 1u);
    unsigned short h = (unsigned short)(r >> 16);
    return __builtin_bit_cast(bf16_t, h);
}

static __device__ __forceinline__ v8f wmma_bf16(v16bf a, v16bf b, v8f c) {
    return __builtin_amdgcn_wmma_f32_16x16x32_bf16(
        /*neg_a=*/false, a, /*neg_b=*/false, b,
        /*c_mod=*/(short)0, c, /*reuse_a=*/false, /*reuse_b=*/false);
}

// Load one 16x32 bf16 operand tile (A layout; same pattern serves as B operand
// when the source is stored so B-matrix columns are contiguous rows).
// ISA layout (16-bit, 16x32): lane<16 row=lane holds K 0..7 & 16..23,
// lane>=16 row=lane-16 holds K 8..15 & 24..31.
static __device__ __forceinline__ v16bf load_bf16_tile(const bf16_t* __restrict__ p,
                                                       int row0, int ld, int k0, int lane) {
    int r = lane & 15, half = lane >> 4;
    const bf16_t* q = p + (size_t)(row0 + r) * (size_t)ld + (size_t)(k0 + half * 8);
    v8bf lo = *(const v8bf*)q;            // 16B
    v8bf hi = *(const v8bf*)(q + 16);     // 16B
    return __builtin_shufflevector(lo, hi, 0,1,2,3,4,5,6,7,8,9,10,11,12,13,14,15);
}

// Shared 64x64 wave-tile GEMM inner loop.
// All 16 b128 loads (4 A tiles + 4 B tiles) are issued before the 16 WMMAs so
// the compiler can stagger s_wait_loadcnt across the whole WMMA burst instead
// of stalling on a recycled B register pair.
template <int KTOT>
static __device__ __forceinline__ void gemm_64x64(const bf16_t* __restrict__ A, int lda, int m0,
                                                  const bf16_t* __restrict__ Bm, int ldb, int n0,
                                                  int lane, v8f acc[4][4]) {
    const int r = lane & 15;
    for (int k0 = 0; k0 < KTOT; k0 += 32) {
        __builtin_prefetch(A  + (size_t)(m0 + r) * lda + k0 + 64, 0, 3);
        __builtin_prefetch(Bm + (size_t)(n0 + r) * ldb + k0 + 64, 0, 3);
        v16bf a[4], b[4];
#pragma unroll
        for (int mi = 0; mi < 4; ++mi)
            a[mi] = load_bf16_tile(A, m0 + mi * 16, lda, k0, lane);
#pragma unroll
        for (int ni = 0; ni < 4; ++ni)
            b[ni] = load_bf16_tile(Bm, n0 + ni * 16, ldb, k0, lane);
#pragma unroll
        for (int ni = 0; ni < 4; ++ni)
#pragma unroll
            for (int mi = 0; mi < 4; ++mi)
                acc[mi][ni] = wmma_bf16(a[mi], b[ni], acc[mi][ni]);
    }
}

// ---------------------------------------------------------------------------
// Kernel 1: elementwise f32 -> bf16 convert (4 elements / thread)
// ---------------------------------------------------------------------------
__global__ __launch_bounds__(256) void xconv_kernel(const float* __restrict__ X,
                                                    bf16_t* __restrict__ Y) {
    int idx = (blockIdx.x * 256 + threadIdx.x) * 4;
    v4f v = *(const v4f*)(X + idx);
    v4bf o;
#pragma unroll
    for (int i = 0; i < 4; ++i) o[i] = f2bf(v[i]);
    *(v4bf*)(Y + idx) = o;
}

// ---------------------------------------------------------------------------
// Kernel 2: weight convert + transpose: Wt[n*D + k] = bf16(W[k*D + n])
// ---------------------------------------------------------------------------
__global__ __launch_bounds__(256) void wconv_kernel(const float* __restrict__ W,
                                                    bf16_t* __restrict__ Wt) {
    int idx = blockIdx.x * 256 + threadIdx.x;     // over D*D
    int k = idx >> 10;
    int n = idx & (DIM - 1);
    Wt[(size_t)n * DIM + k] = f2bf(W[idx]);
}

// ---------------------------------------------------------------------------
// Kernel 3: projection  Y = Xb @ W + b   (Xb: (B*L, D) bf16, Wt: (N,K) bf16)
// transposed==0: Y row-major bf16 (B*L, D)
// transposed==1: Y = V^T per batch, bf16 (B, D, L)
// ---------------------------------------------------------------------------
__global__ __launch_bounds__(128) void proj_kernel(const bf16_t* __restrict__ Xb,
                                                   const bf16_t* __restrict__ Wt,
                                                   const float* __restrict__ bias,
                                                   bf16_t* __restrict__ Y,
                                                   int transposed) {
    const int lane = threadIdx.x & 31;
    const int r = lane & 15, half = lane >> 4;
    const int w = blockIdx.x * 4 + (threadIdx.x >> 5);   // 4096 waves
    const int tm = w >> 4;            // 256 row tiles of 64
    const int tn = w & 15;            // 16 col tiles of 64
    const int m0 = tm * 64, n0 = tn * 64;

    v8f acc[4][4];
#pragma unroll
    for (int ni = 0; ni < 4; ++ni) {
        float bv = bias[n0 + ni * 16 + r];
        v8f c;
#pragma unroll
        for (int i = 0; i < 8; ++i) c[i] = bv;
#pragma unroll
        for (int mi = 0; mi < 4; ++mi) acc[mi][ni] = c;
    }

    gemm_64x64<DIM>(Xb, DIM, m0, Wt, DIM, n0, lane, acc);

#pragma unroll
    for (int mi = 0; mi < 4; ++mi) {
#pragma unroll
        for (int ni = 0; ni < 4; ++ni) {
            int N = n0 + ni * 16 + r;
#pragma unroll
            for (int i = 0; i < 8; ++i) {
                int M = m0 + mi * 16 + i + 8 * half;           // global B*L index
                bf16_t val = f2bf(acc[mi][ni][i]);
                if (transposed) {
                    int bb = M >> 11;          // / SEQ
                    int l  = M & (SEQ - 1);
                    Y[((size_t)bb * DIM + N) * SEQ + l] = val;
                } else {
                    Y[(size_t)M * DIM + N] = val;
                }
            }
        }
    }
}

// ---------------------------------------------------------------------------
// Kernel 4: scores  S[b,q,n] = Q[b,q,:] . K[b,n,:]   (f32 out)
// K row-major rows serve directly as contiguous B-operand columns.
// ---------------------------------------------------------------------------
__global__ __launch_bounds__(128) void scores_kernel(const bf16_t* __restrict__ Qb,
                                                     const bf16_t* __restrict__ Kb,
                                                     float* __restrict__ S) {
    const int lane = threadIdx.x & 31;
    const int r = lane & 15, half = lane >> 4;
    const int w = blockIdx.x * 4 + (threadIdx.x >> 5);   // 8192 waves
    const int batch = w >> 10;                 // 1024 wave-tiles per batch
    const int t = w & 1023;
    const int tm = t >> 5;                     // 32 row tiles of 64
    const int tn = t & 31;                     // 32 col tiles of 64
    const int m0 = tm * 64, n0 = tn * 64;

    const bf16_t* A  = Qb + (size_t)batch * SEQ * DIM;
    const bf16_t* Bm = Kb + (size_t)batch * SEQ * DIM;
    float* out = S + (size_t)batch * SEQ * SEQ;

    v8f acc[4][4];
#pragma unroll
    for (int mi = 0; mi < 4; ++mi)
#pragma unroll
        for (int ni = 0; ni < 4; ++ni) acc[mi][ni] = (v8f)(0.0f);

    gemm_64x64<DIM>(A, DIM, m0, Bm, DIM, n0, lane, acc);

#pragma unroll
    for (int mi = 0; mi < 4; ++mi)
#pragma unroll
        for (int ni = 0; ni < 4; ++ni) {
            int N = n0 + ni * 16 + r;
#pragma unroll
            for (int i = 0; i < 8; ++i) {
                int M = m0 + mi * 16 + i + 8 * half;
                out[(size_t)M * SEQ + N] = acc[mi][ni][i];
            }
        }
}

// ---------------------------------------------------------------------------
// Kernel 5: row softmax over SEQ f32 -> bf16 P.  One 256-thread block per row.
// ---------------------------------------------------------------------------
__global__ __launch_bounds__(256) void softmax_kernel(const float* __restrict__ S,
                                                      bf16_t* __restrict__ P) {
    __shared__ float red[256];
    const int row = blockIdx.x;                  // B*L rows
    const int t = threadIdx.x;
    const float* s = S + (size_t)row * SEQ;
    bf16_t* p = P + (size_t)row * SEQ;

    float v[8];
    float m = -3.4e38f;
#pragma unroll
    for (int j = 0; j < 8; ++j) {
        v[j] = s[t + 256 * j];
        m = fmaxf(m, v[j]);
    }
    red[t] = m; __syncthreads();
    for (int off = 128; off > 0; off >>= 1) {
        if (t < off) red[t] = fmaxf(red[t], red[t + off]);
        __syncthreads();
    }
    m = red[0]; __syncthreads();

    float sum = 0.0f;
#pragma unroll
    for (int j = 0; j < 8; ++j) {
        v[j] = __expf(v[j] - m);
        sum += v[j];
    }
    red[t] = sum; __syncthreads();
    for (int off = 128; off > 0; off >>= 1) {
        if (t < off) red[t] += red[t + off];
        __syncthreads();
    }
    float inv = 1.0f / red[0];
#pragma unroll
    for (int j = 0; j < 8; ++j) p[t + 256 * j] = f2bf(v[j] * inv);
}

// ---------------------------------------------------------------------------
// Kernel 6: output  O[b,q,d] = sum_k P[b,q,k] * V[b,k,d], using Vt (B,D,L)
// so B-operand columns (fixed d, contiguous k) are contiguous in memory.
// ---------------------------------------------------------------------------
__global__ __launch_bounds__(128) void outgemm_kernel(const bf16_t* __restrict__ P,
                                                      const bf16_t* __restrict__ Vt,
                                                      float* __restrict__ O) {
    const int lane = threadIdx.x & 31;
    const int r = lane & 15, half = lane >> 4;
    const int w = blockIdx.x * 4 + (threadIdx.x >> 5);   // 4096 waves
    const int batch = w >> 9;                  // 512 wave-tiles per batch
    const int t = w & 511;
    const int tm = t >> 4;                     // 32 row tiles of 64
    const int tn = t & 15;                     // 16 col tiles of 64
    const int m0 = tm * 64, n0 = tn * 64;

    const bf16_t* A  = P  + (size_t)batch * SEQ * SEQ;
    const bf16_t* Bm = Vt + (size_t)batch * DIM * SEQ;
    float* out = O + (size_t)batch * SEQ * DIM;

    v8f acc[4][4];
#pragma unroll
    for (int mi = 0; mi < 4; ++mi)
#pragma unroll
        for (int ni = 0; ni < 4; ++ni) acc[mi][ni] = (v8f)(0.0f);

    gemm_64x64<SEQ>(A, SEQ, m0, Bm, SEQ, n0, lane, acc);

#pragma unroll
    for (int mi = 0; mi < 4; ++mi)
#pragma unroll
        for (int ni = 0; ni < 4; ++ni) {
            int N = n0 + ni * 16 + r;
#pragma unroll
            for (int i = 0; i < 8; ++i) {
                int M = m0 + mi * 16 + i + 8 * half;
                out[(size_t)M * DIM + N] = acc[mi][ni][i];
            }
        }
}

// ---------------------------------------------------------------------------
// Host-side launcher
// ---------------------------------------------------------------------------
extern "C" void kernel_launch(void* const* d_in, const int* in_sizes, int n_in,
                              void* d_out, int out_size, void* d_ws, size_t ws_size,
                              hipStream_t stream) {
    const float* Xq = (const float*)d_in[0];
    const float* Xk = (const float*)d_in[1];
    const float* Xv = (const float*)d_in[2];
    const float* Wq = (const float*)d_in[3];
    const float* bq = (const float*)d_in[4];
    const float* Wk = (const float*)d_in[5];
    const float* bk = (const float*)d_in[6];
    const float* Wv = (const float*)d_in[7];
    const float* bv = (const float*)d_in[8];

    char* ws = (char*)d_ws;
    const size_t MB = 1ull << 20;
    bf16_t* WtQ = (bf16_t*)(ws + 0 * MB);     //  2 MiB
    bf16_t* WtK = (bf16_t*)(ws + 2 * MB);     //  2 MiB
    bf16_t* WtV = (bf16_t*)(ws + 4 * MB);     //  2 MiB
    bf16_t* Xqb = (bf16_t*)(ws + 6 * MB);     // 32 MiB  (B,L,D) bf16
    bf16_t* Xkb = (bf16_t*)(ws + 38 * MB);    // 32 MiB
    bf16_t* Xvb = (bf16_t*)(ws + 70 * MB);    // 32 MiB
    bf16_t* Qb  = (bf16_t*)(ws + 102 * MB);   // 32 MiB  (B,L,D) bf16
    bf16_t* Kb  = (bf16_t*)(ws + 134 * MB);   // 32 MiB  (B,L,D) bf16
    bf16_t* Vt  = (bf16_t*)(ws + 166 * MB);   // 32 MiB  (B,D,L) bf16
    float*  S   = (float*) (ws + 198 * MB);   // 128 MiB (B,L,L) f32
    bf16_t* P   = (bf16_t*)(ws + 326 * MB);   // 64 MiB  (B,L,L) bf16
    // total workspace: 390 MiB

    const int NX = BATCH * SEQ * DIM;         // 16,777,216 elements

    // 1) activation convert (once), weight transpose+convert
    xconv_kernel<<<NX / 1024, 256, 0, stream>>>(Xq, Xqb);
    xconv_kernel<<<NX / 1024, 256, 0, stream>>>(Xk, Xkb);
    xconv_kernel<<<NX / 1024, 256, 0, stream>>>(Xv, Xvb);
    wconv_kernel<<<(DIM * DIM) / 256, 256, 0, stream>>>(Wq, WtQ);
    wconv_kernel<<<(DIM * DIM) / 256, 256, 0, stream>>>(Wk, WtK);
    wconv_kernel<<<(DIM * DIM) / 256, 256, 0, stream>>>(Wv, WtV);

    // 2) projections: 4096 waves / 4 per block
    proj_kernel<<<1024, 128, 0, stream>>>(Xqb, WtQ, bq, Qb, 0);
    proj_kernel<<<1024, 128, 0, stream>>>(Xkb, WtK, bk, Kb, 0);
    proj_kernel<<<1024, 128, 0, stream>>>(Xvb, WtV, bv, Vt, 1);

    // 3) scores: 8192 waves / 4 per block
    scores_kernel<<<2048, 128, 0, stream>>>(Qb, Kb, S);

    // 4) softmax: one block per row
    softmax_kernel<<<BATCH * SEQ, 256, 0, stream>>>(S, P);

    // 5) output GEMM: 4096 waves / 4 per block
    outgemm_kernel<<<1024, 128, 0, stream>>>(P, Vt, (float*)d_out);
}
